// EdgeRegressionModel_23570780521007
// MI455X (gfx1250) — matrix operand order repaired and checked by reading, not compiled, not run on hardware
//
#include <hip/hip_runtime.h>
#include <hip/hip_bf16.h>

#define NN 25000
#define EE 250000
#define GG 64
#define WSTRIDE 1028   // padded LDS row stride (floats); 8*1028 % 64 == 32 -> bank-safe

typedef __attribute__((ext_vector_type(16))) __bf16 v16bf;
typedef __attribute__((ext_vector_type(8)))  float  v8f;

__device__ __forceinline__ unsigned short f2bf(float f) {
    union { float f; unsigned u; } v; v.f = f;
    unsigned r = v.u + 0x7FFFu + ((v.u >> 16) & 1u);
    return (unsigned short)(r >> 16);
}
__device__ __forceinline__ float lrelu(float x) { return x < 0.f ? 0.01f * x : x; }

// ---------------- column-normalization (_norm_cols) ----------------
__global__ __launch_bounds__(256) void k_nc_stats(const float* __restrict__ A, int rows, int cols,
                                                  int c0, int c1, int c2, int c3, int c4, int K,
                                                  float* __restrict__ acc) {
    int cl[5] = {c0, c1, c2, c3, c4};
    float s[5] = {0, 0, 0, 0, 0}, q[5] = {0, 0, 0, 0, 0};
    for (int r = blockIdx.x * blockDim.x + threadIdx.x; r < rows; r += gridDim.x * blockDim.x)
        for (int k = 0; k < K; ++k) {
            float v = A[(size_t)r * cols + cl[k]];
            s[k] += v; q[k] += v * v;
        }
    __shared__ float red[256];
    for (int k = 0; k < K; ++k) {
        red[threadIdx.x] = s[k]; __syncthreads();
        for (int st = 128; st > 0; st >>= 1) { if (threadIdx.x < st) red[threadIdx.x] += red[threadIdx.x + st]; __syncthreads(); }
        if (threadIdx.x == 0) atomicAdd(&acc[k], red[0]);
        __syncthreads();
        red[threadIdx.x] = q[k]; __syncthreads();
        for (int st = 128; st > 0; st >>= 1) { if (threadIdx.x < st) red[threadIdx.x] += red[threadIdx.x + st]; __syncthreads(); }
        if (threadIdx.x == 0) atomicAdd(&acc[8 + k], red[0]);
        __syncthreads();
    }
}

__global__ void k_nc_apply(const float* __restrict__ A, float* __restrict__ O, int rows, int cols,
                           int c0, int c1, int c2, int c3, int c4, int K,
                           const float* __restrict__ acc) {
    int cl[5] = {c0, c1, c2, c3, c4};
    size_t total = (size_t)rows * cols;
    for (size_t p = blockIdx.x * (size_t)blockDim.x + threadIdx.x; p < total; p += (size_t)gridDim.x * blockDim.x) {
        int c = (int)(p % cols);
        float v = A[p];
        for (int k = 0; k < K; ++k)
            if (c == cl[k]) {
                float m  = acc[k] / rows;
                float vr = (acc[8 + k] - rows * m * m) / (rows - 1);   // ddof=1
                float sd = sqrtf(fmaxf(vr, 0.f));
                v = (v - m) / (sd + 1e-8f);
            }
        O[p] = v;
    }
}

// ---------------- generic linear Y = X @ W^T + b ----------------
__global__ void k_linear(const float* __restrict__ X, const float* __restrict__ W,
                         const float* __restrict__ b, float* __restrict__ Y,
                         int rows, int din, int dout) {
    size_t total = (size_t)rows * dout;
    for (size_t p = blockIdx.x * (size_t)blockDim.x + threadIdx.x; p < total; p += (size_t)gridDim.x * blockDim.x) {
        int j = (int)(p % dout);
        size_t r = p / dout;
        const float* xr = X + r * din;
        const float* wr = W + (size_t)j * din;
        float s = b[j];
        for (int i = 0; i < din; ++i) s += xr[i] * wr[i];
        Y[p] = s;
    }
}

// ---------------- BN column stats (32 cols) ----------------
__global__ __launch_bounds__(256) void k_colstats(const float* __restrict__ A, int rows,
                                                  float* __restrict__ acc) {
    const int t = threadIdx.x, col = t & 31, grp = t >> 5;
    float s = 0.f, q = 0.f;
    for (int r = blockIdx.x * 8 + grp; r < rows; r += gridDim.x * 8) {
        float v = A[(size_t)r * 32 + col];
        s += v; q += v * v;
    }
    __shared__ float red[256];
    red[t] = s; __syncthreads();
    if (t < 32) { float a = 0.f; for (int k = 0; k < 8; ++k) a += red[t + k * 32]; atomicAdd(&acc[t], a); }
    __syncthreads();
    red[t] = q; __syncthreads();
    if (t < 32) { float a = 0.f; for (int k = 0; k < 8; ++k) a += red[t + k * 32]; atomicAdd(&acc[32 + t], a); }
}

// ------- BN apply + leaky relu; optional bf16 mirror PRE-SWIZZLED into WMMA A layout -------
// A-fragment (16-bit A 16x32, ISA 7.12.2): for channel k of edge e:
//   hlf = (k>>3)&1 ; kk = k-8*hlf ; r = kk<8 ? kk>>1 : 4+((kk-16)>>1) ; slot = 2r+(kk&1)
//   dest = ((e>>4)*32 + (e&15) + 16*hlf)*16 + slot       (16 contiguous ushorts per lane)
__global__ void k_bn(float* __restrict__ Y, int rows, const float* __restrict__ acc,
                     const float* __restrict__ g, const float* __restrict__ beta,
                     unsigned short* __restrict__ bfout) {
    size_t total = (size_t)rows * 32;
    float inv = 1.0f / rows;
    for (size_t p = blockIdx.x * (size_t)blockDim.x + threadIdx.x; p < total; p += (size_t)gridDim.x * blockDim.x) {
        int c = (int)(p & 31);
        float m  = acc[c] * inv;
        float vr = acc[32 + c] * inv - m * m;
        float sc = g[c] * rsqrtf(vr + 1e-5f);
        float v  = lrelu((Y[p] - m) * sc + beta[c]);
        Y[p] = v;
        if (bfout) {
            size_t e = p >> 5;
            int k    = c;
            int hlf  = (k >> 3) & 1;
            int kk   = k - 8 * hlf;
            int r    = (kk < 8) ? (kk >> 1) : (4 + ((kk - 16) >> 1));
            int slot = 2 * r + (kk & 1);
            size_t idx = ((e >> 4) * 32 + ((e & 15) + 16 * hlf)) * 16 + slot;
            bfout[idx] = f2bf(v);
        }
    }
}

// ---------------- Gram matrix S = H^T H (32x32) + colsum u ----------------
__global__ __launch_bounds__(256) void k_gram(const float* __restrict__ H, int rows,
                                              float* __restrict__ S, float* __restrict__ u) {
    __shared__ float rb[8 * 32];
    const int t = threadIdx.x;
    const int i0 = (t * 4) >> 5, k0 = (t * 4) & 31;
    float sl0 = 0, sl1 = 0, sl2 = 0, sl3 = 0, ul = 0;
    for (int r0 = blockIdx.x * 8; r0 < rows; r0 += gridDim.x * 8) {
        int rr_ = r0 + (t >> 5);
        rb[t] = (rr_ < rows) ? H[(size_t)rr_ * 32 + (t & 31)] : 0.f;
        __syncthreads();
#pragma unroll
        for (int rr = 0; rr < 8; ++rr) {
            float hi = rb[rr * 32 + i0];
            sl0 += hi * rb[rr * 32 + k0];
            sl1 += hi * rb[rr * 32 + k0 + 1];
            sl2 += hi * rb[rr * 32 + k0 + 2];
            sl3 += hi * rb[rr * 32 + k0 + 3];
        }
        if (t < 32) { for (int rr = 0; rr < 8; ++rr) ul += rb[rr * 32 + t]; }
        __syncthreads();
    }
    atomicAdd(&S[t * 4 + 0], sl0);
    atomicAdd(&S[t * 4 + 1], sl1);
    atomicAdd(&S[t * 4 + 2], sl2);
    atomicAdd(&S[t * 4 + 3], sl3);
    if (t < 32) atomicAdd(&u[t], ul);
}

// -------- BN2 scale/shift for the 1024-wide BN, from 32x32 Gram (no E-pass) --------
__global__ __launch_bounds__(1024) void k_bn2stats(const float* __restrict__ S, const float* __restrict__ u,
                                                   const float* __restrict__ W2, const float* __restrict__ b2,
                                                   const float* __restrict__ g2, const float* __restrict__ bb2,
                                                   float rowsInv, float* __restrict__ sc2, float* __restrict__ sh2) {
    __shared__ float ub[32];
    __shared__ float C[1024];
    const int t = threadIdx.x;
    if (t < 32) ub[t] = u[t] * rowsInv;
    __syncthreads();
    { int i = t >> 5, k = t & 31; C[t] = S[t] * rowsInv - ub[i] * ub[k]; }
    __syncthreads();
    float w[32];
#pragma unroll
    for (int i = 0; i < 32; ++i) w[i] = W2[t * 32 + i];
    float mean = b2[t], var = 0.f;
#pragma unroll
    for (int i = 0; i < 32; ++i) {
        mean += w[i] * ub[i];
        float vi = 0.f;
        for (int k = 0; k < 32; ++k) vi += C[i * 32 + k] * w[k];
        var += w[i] * vi;
    }
    var = fmaxf(var, 0.f);
    float sc = g2[t] * rsqrtf(var + 1e-5f);
    sc2[t] = sc;
    sh2[t] = bb2[t] - mean * sc;
}

// -------- pack W2 into B-fragment bf16 layout, with BN scale folded in --------
// B 16-bit 32x16: lanes 0-15 hold K=0..7,16..23, lanes 16-31 hold K=8..15,24..31; n = lane&15.
__global__ void k_packw2(const float* __restrict__ W2, const float* __restrict__ sc2,
                         unsigned short* __restrict__ w2bf) {
    int idx = blockIdx.x * blockDim.x + threadIdx.x;
    if (idx >= 64 * 32 * 16) return;
    int t = idx >> 9, l = (idx >> 4) & 31, s = idx & 15;
    int n = l & 15, k = (l >> 4) * 8 + (s & 7) + (s >> 3) * 16;
    int j = t * 16 + n;
    w2bf[idx] = f2bf(W2[j * 32 + k] * sc2[j]);
}

// ---------------- fused NNConv message kernel (bf16 WMMA, lean epilogue) ----------------
// 16 edges/block. acc initialized with BN shift (C operand), BN scale folded into B,
// lrelu = max(x, 0.01x), f32 w tile staged in dynamic LDS, then per-edge matvec+scatter.
__global__ __launch_bounds__(256) void k_nnconv_msg(const unsigned short* __restrict__ h1sw,
                                                    const unsigned short* __restrict__ w2bf,
                                                    const float* __restrict__ sh2,
                                                    const float* __restrict__ xcur,
                                                    const int* __restrict__ src,
                                                    const int* __restrict__ dst,
                                                    float* __restrict__ agg) {
    extern __shared__ float wsh[];          // 16 * WSTRIDE floats (~64 KB, dynamic)
    __shared__ float xs[16 * 32];
    __shared__ int sidx[16];
    __shared__ int didx[16];

    const int tid  = threadIdx.x;
    const int lane = tid & 31;
    const int wv   = tid >> 5;

    if (tid < 16) { sidx[tid] = src[blockIdx.x * 16 + tid]; didx[tid] = dst[blockIdx.x * 16 + tid]; }

    // A fragment: 16 contiguous bf16 per lane (pre-swizzled by k_bn)
    const unsigned short* ap = h1sw + ((size_t)blockIdx.x * 32 + lane) * 16;
    v16bf afrag;
    __builtin_memcpy(&afrag, ap, 32);

    const unsigned short* bbase = w2bf + ((size_t)(wv * 8) * 32 + lane) * 16;
    __builtin_prefetch(bbase, 0, 1);        // global_prefetch_b8 on the B stream

    __syncthreads();
    for (int p = tid; p < 512; p += 256) {
        int m = p >> 5, i = p & 31;
        xs[p] = xcur[(size_t)sidx[m] * 32 + i];
    }

    const int nIdx = lane & 15;
    const int hlf  = lane >> 4;

    v16bf bcur;
    __builtin_memcpy(&bcur, bbase, 32);
#pragma unroll
    for (int ti = 0; ti < 8; ++ti) {
        v16bf bnext;
        if (ti < 7) __builtin_memcpy(&bnext, bbase + (size_t)(ti + 1) * 512, 32);
        const int j  = (wv * 8 + ti) * 16 + nIdx;
        const float sh = sh2[j];
        v8f acc = {sh, sh, sh, sh, sh, sh, sh, sh};   // BN shift via C operand
        acc = __builtin_amdgcn_wmma_f32_16x16x32_bf16(false, afrag, false, bcur,
                                                      (short)0, acc, false, false);
#pragma unroll
        for (int r = 0; r < 8; ++r) {
            int m = (hlf == 0) ? r : (r + 8);          // C/D: lanes0-15 M=r, lanes16-31 M=r+8
            float z = acc[r];
            wsh[m * WSTRIDE + j] = fmaxf(z, 0.01f * z);   // leaky relu, slope<1
        }
        if (ti < 7) bcur = bnext;
    }
    __syncthreads();

    for (int p = tid; p < 512; p += 256) {
        int m = p >> 5, o = p & 31;
        float s = 0.f;
#pragma unroll
        for (int i = 0; i < 32; ++i)
            s += xs[m * 32 + i] * wsh[m * WSTRIDE + i * 32 + o];
        atomicAdd(&agg[(size_t)didx[m] * 32 + o], s);
    }
}

// ---------------- root transform + bias + l2 normalization ----------------
__global__ void k_root_l2(const float* __restrict__ agg, const float* __restrict__ xin,
                          const float* __restrict__ root, const float* __restrict__ bias,
                          float* __restrict__ xout, int n) {
    int v = blockIdx.x * blockDim.x + threadIdx.x;
    if (v >= n) return;
    float xr[32], y[32];
#pragma unroll
    for (int i = 0; i < 32; ++i) xr[i] = xin[(size_t)v * 32 + i];
    float nrm = 0.f;
#pragma unroll
    for (int o = 0; o < 32; ++o) {
        float s = agg[(size_t)v * 32 + o] + bias[o];
        for (int i = 0; i < 32; ++i) s += xr[i] * root[o * 32 + i];
        y[o] = s;
        nrm += s * s;
    }
    float inv = 1.f / fmaxf(sqrtf(nrm), 1e-12f);
#pragma unroll
    for (int o = 0; o < 32; ++o) xout[(size_t)v * 32 + o] = y[o] * inv;
}

// ---------------- regressor first layer on concat(x[src], x[dst]) ----------------
__global__ void k_reg_lin1(const float* __restrict__ x2, const int* __restrict__ src,
                           const int* __restrict__ dst, const float* __restrict__ W,
                           const float* __restrict__ b, float* __restrict__ Y, int E_) {
    size_t total = (size_t)E_ * 32;
    for (size_t p = blockIdx.x * (size_t)blockDim.x + threadIdx.x; p < total; p += (size_t)gridDim.x * blockDim.x) {
        int j = (int)(p & 31);
        size_t e = p >> 5;
        const float* xsrc = x2 + (size_t)src[e] * 32;
        const float* xdst = x2 + (size_t)dst[e] * 32;
        const float* wr = W + (size_t)j * 64;
        float s = b[j];
        for (int i = 0; i < 32; ++i) s += xsrc[i] * wr[i];
        for (int i = 0; i < 32; ++i) s += xdst[i] * wr[32 + i];
        Y[p] = s;
    }
}

__global__ void k_lin3(const float* __restrict__ Z, const float* __restrict__ W,
                       const float* __restrict__ b, float* __restrict__ out, int E_) {
    int e = blockIdx.x * blockDim.x + threadIdx.x;
    if (e >= E_) return;
    float s = b[0];
    for (int i = 0; i < 32; ++i) s += Z[(size_t)e * 32 + i] * W[i];
    out[e] = s;
}

// ---------------- graph pooling (segment sum + counts) ----------------
__global__ void k_seg(const float* __restrict__ xg, const int* __restrict__ batch,
                      float* __restrict__ gsum, float* __restrict__ cnt, int n) {
    size_t total = (size_t)n * 32;
    for (size_t p = blockIdx.x * (size_t)blockDim.x + threadIdx.x; p < total; p += (size_t)gridDim.x * blockDim.x) {
        int c = (int)(p & 31);
        size_t v = p >> 5;
        int b = batch[v];
        atomicAdd(&gsum[(size_t)b * 32 + c], xg[p]);
        if (c == 0) atomicAdd(&cnt[b], 1.0f);
    }
}

// ---------------- decode head (single block; tiny) ----------------
__global__ __launch_bounds__(256) void k_decode(const float* __restrict__ gsum, const float* __restrict__ cnt,
                                                const float* __restrict__ W1, const float* __restrict__ b1,
                                                const float* __restrict__ W2, const float* __restrict__ b2,
                                                float* __restrict__ out) {
    __shared__ float m[2048];
    __shared__ float h[2048];
    int t = threadIdx.x;
    for (int p = t; p < 2048; p += 256) m[p] = gsum[p] / fmaxf(cnt[p >> 5], 1.0f);
    __syncthreads();
    for (int p = t; p < 2048; p += 256) {
        int g = p >> 5, j = p & 31;
        float s = b1[j];
        for (int c = 0; c < 32; ++c) s += m[g * 32 + c] * W1[j * 32 + c];
        h[p] = lrelu(s);
    }
    __syncthreads();
    for (int p = t; p < 2048; p += 256) {
        int g = p >> 5, j = p & 31;
        float s = b2[j];
        for (int c = 0; c < 32; ++c) s += h[g * 32 + c] * W2[j * 32 + c];
        out[p] = s;
    }
}

// =====================================================================
extern "C" void kernel_launch(void* const* d_in, const int* in_sizes, int n_in,
                              void* d_out, int out_size, void* d_ws, size_t ws_size,
                              hipStream_t stream) {
    (void)in_sizes; (void)n_in; (void)out_size; (void)ws_size;
    auto F = [&](int i) { return (const float*)d_in[i]; };

    const float* x     = F(0);
    const int*   eidx  = (const int*)d_in[1];
    const float* eattr = F(2);
    const int*   batch = (const int*)d_in[3];
    const int*   srcI  = eidx;
    const int*   dstI  = eidx + EE;

    // params in dict order
    const float *ne_l1W = F(4),  *ne_l1b = F(5),  *ne_g1 = F(6),  *ne_b1 = F(7);
    const float *ne_l2W = F(8),  *ne_l2b = F(9),  *ne_g2 = F(10), *ne_b2 = F(11);
    const float *ee_l1W = F(12), *ee_l1b = F(13), *ee_g1 = F(14), *ee_b1 = F(15);
    const float *ee_l2W = F(16), *ee_l2b = F(17), *ee_g2 = F(18), *ee_b2 = F(19);
    const float *c1_l1W = F(20), *c1_l1b = F(21), *c1_g1 = F(22), *c1_b1 = F(23);
    const float *c1_l2W = F(24), *c1_l2b = F(25), *c1_g2 = F(26), *c1_b2 = F(27);
    const float *c1_root = F(28), *c1_bias = F(29);
    const float *c2_l1W = F(30), *c2_l1b = F(31), *c2_g1 = F(32), *c2_b1 = F(33);
    const float *c2_l2W = F(34), *c2_l2b = F(35), *c2_g2 = F(36), *c2_b2 = F(37);
    const float *c2_root = F(38), *c2_bias = F(39);
    const float *r_l1W = F(40), *r_l1b = F(41), *r_g1 = F(42), *r_b1 = F(43);
    const float *r_l2W = F(44), *r_l2b = F(45), *r_g2 = F(46), *r_b2 = F(47);
    const float *r_l3W = F(48), *r_l3b = F(49);
    const float *g_l1W = F(50), *g_l1b = F(51), *g_g1 = F(52), *g_b1 = F(53);
    const float *g_l2W = F(54), *g_l2b = F(55), *g_g2 = F(56), *g_b2 = F(57);
    const float *d_l1W = F(58), *d_l1b = F(59), *d_l2W = F(60), *d_l2b = F(61);

    // workspace layout (floats)
    float* ws = (float*)d_ws;
    size_t off = 0;
    float* x_n  = ws + off; off += (size_t)NN * 16;
    float* ean  = ws + off; off += (size_t)EE * 16;
    float* ea   = ws + off; off += (size_t)EE * 32;
    float* h1   = ws + off; off += (size_t)EE * 32;
    unsigned short* h1sw = (unsigned short*)(ws + off); off += (size_t)EE * 16;  // E*32 ushorts, swizzled
    float* xA   = ws + off; off += (size_t)NN * 32;
    float* xB   = ws + off; off += (size_t)NN * 32;
    float* aggb = ws + off; off += (size_t)NN * 32;
    float* xg   = ws + off; off += (size_t)NN * 32;
    float* acc  = ws + off; off += 64;
    float* Sg   = ws + off; off += 1024;
    float* ug   = ws + off; off += 32;
    float* sc2  = ws + off; off += 1024;
    float* sh2  = ws + off; off += 1024;
    float* nc   = ws + off; off += 16;
    float* gsum = ws + off; off += 2048;
    float* cntf = ws + off; off += 64;
    unsigned short* w2bf = (unsigned short*)(ws + off); off += 16384;

    float* out_scores = (float*)d_out;        // E edge scores
    float* out_ge     = out_scores + EE;      // 64 x 32 graph embeddings

    auto GRD = [](size_t t) { return (unsigned)((t + 255) / 256); };
    auto MS  = [&](void* p, size_t bytes) { hipMemsetAsync(p, 0, bytes, stream); };

    // ---- norm_cols(x, {0,6,7}) and norm_cols(edge_attr, {0,2,7,8,9}) ----
    MS(nc, 16 * 4);
    k_nc_stats<<<256, 256, 0, stream>>>(x, NN, 16, 0, 6, 7, 0, 0, 3, nc);
    k_nc_apply<<<GRD((size_t)NN * 16), 256, 0, stream>>>(x, x_n, NN, 16, 0, 6, 7, 0, 0, 3, nc);
    MS(nc, 16 * 4);
    k_nc_stats<<<256, 256, 0, stream>>>(eattr, EE, 16, 0, 2, 7, 8, 9, 5, nc);
    k_nc_apply<<<GRD((size_t)EE * 16), 256, 0, stream>>>(eattr, ean, EE, 16, 0, 2, 7, 8, 9, 5, nc);

    // ---- node encoder -> xA ----
    k_linear<<<GRD((size_t)NN * 32), 256, 0, stream>>>(x_n, ne_l1W, ne_l1b, xB, NN, 16, 32);
    MS(acc, 64 * 4);
    k_colstats<<<256, 256, 0, stream>>>(xB, NN, acc);
    k_bn<<<GRD((size_t)NN * 32), 256, 0, stream>>>(xB, NN, acc, ne_g1, ne_b1, (unsigned short*)nullptr);
    k_linear<<<GRD((size_t)NN * 32), 256, 0, stream>>>(xB, ne_l2W, ne_l2b, xA, NN, 32, 32);
    MS(acc, 64 * 4);
    k_colstats<<<256, 256, 0, stream>>>(xA, NN, acc);
    k_bn<<<GRD((size_t)NN * 32), 256, 0, stream>>>(xA, NN, acc, ne_g2, ne_b2, (unsigned short*)nullptr);

    // ---- edge encoder -> ea ----
    k_linear<<<GRD((size_t)EE * 32), 256, 0, stream>>>(ean, ee_l1W, ee_l1b, h1, EE, 16, 32);
    MS(acc, 64 * 4);
    k_colstats<<<256, 256, 0, stream>>>(h1, EE, acc);
    k_bn<<<GRD((size_t)EE * 32), 256, 0, stream>>>(h1, EE, acc, ee_g1, ee_b1, (unsigned short*)nullptr);
    k_linear<<<GRD((size_t)EE * 32), 256, 0, stream>>>(h1, ee_l2W, ee_l2b, ea, EE, 32, 32);
    MS(acc, 64 * 4);
    k_colstats<<<256, 256, 0, stream>>>(ea, EE, acc);
    k_bn<<<GRD((size_t)EE * 32), 256, 0, stream>>>(ea, EE, acc, ee_g2, ee_b2, (unsigned short*)nullptr);

    // ---- NNConv (fused, WMMA) ----
    const size_t msgShmem = (size_t)16 * WSTRIDE * sizeof(float);   // dynamic LDS for w tile
    auto runConv = [&](const float* l1W, const float* l1b, const float* g1, const float* b1,
                       const float* l2W, const float* l2b, const float* g2, const float* b2,
                       const float* root, const float* bias, const float* xin, float* xout) {
        k_linear<<<GRD((size_t)EE * 32), 256, 0, stream>>>(ea, l1W, l1b, h1, EE, 32, 32);
        MS(acc, 64 * 4);
        k_colstats<<<256, 256, 0, stream>>>(h1, EE, acc);
        k_bn<<<GRD((size_t)EE * 32), 256, 0, stream>>>(h1, EE, acc, g1, b1, h1sw);
        MS(Sg, 1024 * 4);
        MS(ug, 32 * 4);
        k_gram<<<512, 256, 0, stream>>>(h1, EE, Sg, ug);
        k_bn2stats<<<1, 1024, 0, stream>>>(Sg, ug, l2W, l2b, g2, b2, 1.0f / EE, sc2, sh2);
        k_packw2<<<128, 256, 0, stream>>>(l2W, sc2, w2bf);
        MS(aggb, (size_t)NN * 32 * 4);
        k_nnconv_msg<<<EE / 16, 256, msgShmem, stream>>>(h1sw, w2bf, sh2, xin, srcI, dstI, aggb);
        k_root_l2<<<(NN + 255) / 256, 256, 0, stream>>>(aggb, xin, root, bias, xout, NN);
    };
    runConv(c1_l1W, c1_l1b, c1_g1, c1_b1, c1_l2W, c1_l2b, c1_g2, c1_b2, c1_root, c1_bias, xA, xB);
    runConv(c2_l1W, c2_l1b, c2_g1, c2_b1, c2_l2W, c2_l2b, c2_g2, c2_b2, c2_root, c2_bias, xB, xA);
    // final node features in xA

    // ---- edge regressor ----
    k_reg_lin1<<<GRD((size_t)EE * 32), 256, 0, stream>>>(xA, srcI, dstI, r_l1W, r_l1b, h1, EE);
    MS(acc, 64 * 4);
    k_colstats<<<256, 256, 0, stream>>>(h1, EE, acc);
    k_bn<<<GRD((size_t)EE * 32), 256, 0, stream>>>(h1, EE, acc, r_g1, r_b1, (unsigned short*)nullptr);
    k_linear<<<GRD((size_t)EE * 32), 256, 0, stream>>>(h1, r_l2W, r_l2b, ea, EE, 32, 32);
    MS(acc, 64 * 4);
    k_colstats<<<256, 256, 0, stream>>>(ea, EE, acc);
    k_bn<<<GRD((size_t)EE * 32), 256, 0, stream>>>(ea, EE, acc, r_g2, r_b2, (unsigned short*)nullptr);
    k_lin3<<<(EE + 255) / 256, 256, 0, stream>>>(ea, r_l3W, r_l3b, out_scores, EE);

    // ---- graph encoder + pooling + decode ----
    k_linear<<<GRD((size_t)NN * 32), 256, 0, stream>>>(xA, g_l1W, g_l1b, xB, NN, 32, 32);
    MS(acc, 64 * 4);
    k_colstats<<<256, 256, 0, stream>>>(xB, NN, acc);
    k_bn<<<GRD((size_t)NN * 32), 256, 0, stream>>>(xB, NN, acc, g_g1, g_b1, (unsigned short*)nullptr);
    k_linear<<<GRD((size_t)NN * 32), 256, 0, stream>>>(xB, g_l2W, g_l2b, xg, NN, 32, 32);
    MS(acc, 64 * 4);
    k_colstats<<<256, 256, 0, stream>>>(xg, NN, acc);
    k_bn<<<GRD((size_t)NN * 32), 256, 0, stream>>>(xg, NN, acc, g_g2, g_b2, (unsigned short*)nullptr);
    MS(gsum, 2048 * 4);
    MS(cntf, 64 * 4);
    k_seg<<<GRD((size_t)NN * 32), 256, 0, stream>>>(xg, batch, gsum, cntf, NN);
    k_decode<<<1, 256, 0, stream>>>(gsum, cntf, d_l1W, d_l1b, d_l2W, d_l2b, out_ge);
}
